// CausalMultiHeadSelfAttention_57990648431449
// MI455X (gfx1250) — compile-verified, matrix operand order checked by reading, not run on hardware
//
#include <hip/hip_runtime.h>
#include <hip/hip_bf16.h>

typedef __bf16 bf16_t;
typedef __attribute__((ext_vector_type(16))) __bf16 v16bf;
typedef __attribute__((ext_vector_type(8)))  float  v8f;

#define D_MODEL 2048
#define NUM_HEADS 16
#define HEAD_DIM 128
#define SEQ 2048
#define BATCH 2
#define MROWS (BATCH * SEQ)          // 4096
#define LN_THETA 9.210340371976184f  // ln(10000)
#define NEG_MASK 10000000.0f
#define SM_SCALE 0.08838834764831845f // 1/sqrt(128)

static __device__ __forceinline__ v8f wmma_bf16(v16bf a, v16bf b, v8f c) {
    // D = A(16x32) * B(32x16) + C, f32 accumulate
    return __builtin_amdgcn_wmma_f32_16x16x32_bf16(
        /*neg_a=*/false, a, /*neg_b=*/false, b,
        /*c_mod=*/(short)0, c, /*reuse_a=*/false, /*reuse_b=*/false);
}

static __device__ __forceinline__ v8f v8f_zero() {
    v8f z = {0.f, 0.f, 0.f, 0.f, 0.f, 0.f, 0.f, 0.f};
    return z;
}

// ---------------------------------------------------------------------------
// fp32 -> bf16 conversion
// ---------------------------------------------------------------------------
__global__ void f32_to_bf16_kernel(const float* __restrict__ in,
                                   bf16_t* __restrict__ out, int n) {
    int i = blockIdx.x * blockDim.x + threadIdx.x;
    if (i < n) out[i] = (bf16_t)in[i];
}

// ---------------------------------------------------------------------------
// GEMM: C[M,N] = A[M,K] * Bt[N,K]^T using v_wmma_f32_16x16x32_bf16.
// Block = 128 threads = 4 waves. Wave tile = 32x64 (2x4 WMMA C-tiles):
// 8 WMMAs per 6 fragment loads per K-step. Block tile = 64x128.
// A/B fragments are contiguous 32B row chunks of row-major A / Bt.
// ---------------------------------------------------------------------------
template <typename OutT>
__global__ __launch_bounds__(128, 1)
void gemm_bf16_wmma(const bf16_t* __restrict__ A,
                    const bf16_t* __restrict__ Bt,
                    OutT* __restrict__ C,
                    int Mdim, int Ndim, int Kdim) {
    const int lane = threadIdx.x & 31;
    const int wave = threadIdx.x >> 5;                 // 0..3
    const int m_blk = blockIdx.y * 64 + (wave & 1) * 32;
    const int n_blk = blockIdx.x * 128 + (wave >> 1) * 64;

    const int row  = lane & 15;          // A row / Bt row within 16-tile
    const int kgrp = (lane >> 4) * 16;   // K sub-chunk select (0 or 16)

    const bf16_t* pA0 = A  + (size_t)(m_blk + row) * Kdim + kgrp;
    const bf16_t* pB0 = Bt + (size_t)(n_blk + row) * Kdim + kgrp;

    v8f acc[2][4];
    #pragma unroll
    for (int i = 0; i < 2; ++i)
        #pragma unroll
        for (int j = 0; j < 4; ++j) acc[i][j] = v8f_zero();

    for (int k0 = 0; k0 < Kdim; k0 += 32) {
        v16bf af[2], bf[4];
        #pragma unroll
        for (int i = 0; i < 2; ++i)
            af[i] = *(const v16bf*)(pA0 + (size_t)(i * 16) * Kdim + k0);
        #pragma unroll
        for (int j = 0; j < 4; ++j)
            bf[j] = *(const v16bf*)(pB0 + (size_t)(j * 16) * Kdim + k0);
        #pragma unroll
        for (int i = 0; i < 2; ++i)
            #pragma unroll
            for (int j = 0; j < 4; ++j)
                acc[i][j] = wmma_bf16(af[i], bf[j], acc[i][j]);
    }

    // C/D fragment layout: lane = n + 16*(m>=8), VGPR r holds row (lane>>4)*8 + r
    const int n_lane  = lane & 15;
    const int m_base8 = (lane >> 4) * 8;
    #pragma unroll
    for (int i = 0; i < 2; ++i)
        #pragma unroll
        for (int j = 0; j < 4; ++j)
            #pragma unroll
            for (int r = 0; r < 8; ++r) {
                int m = m_blk + i * 16 + m_base8 + r;
                int n = n_blk + j * 16 + n_lane;
                C[(size_t)m * Ndim + n] = (OutT)acc[i][j][r];
            }
}

// ---------------------------------------------------------------------------
// RoPE applied in place to Q and K (bf16, [B,S,D] layout); V transposed into
// Vt[b,h,d,s] (bf16) so PV WMMA B-fragments are contiguous.
// One thread per (b, h, s, even/odd pair).
// ---------------------------------------------------------------------------
__global__ void rope_v_kernel(bf16_t* __restrict__ q, bf16_t* __restrict__ k,
                              const bf16_t* __restrict__ v,
                              const int* __restrict__ pos,
                              bf16_t* __restrict__ Vt) {
    int idx = blockIdx.x * blockDim.x + threadIdx.x;
    if (idx >= BATCH * NUM_HEADS * SEQ * (HEAD_DIM / 2)) return;
    const int kp = idx & 63;              // pair index 0..63
    const int s  = (idx >> 6) & (SEQ - 1);
    const int hh = (idx >> 17) & (NUM_HEADS - 1);
    const int b  = idx >> 21;

    const size_t src = ((size_t)b * SEQ + s) * D_MODEL + hh * HEAD_DIM;
    const float p = (float)pos[s];
    const float inv_freq = __expf(-(float)(2 * kp) * LN_THETA * (1.0f / (float)HEAD_DIM));
    float sn, cs;
    __sincosf(p * inv_freq, &sn, &cs);

    float qe = (float)q[src + 2 * kp], qo = (float)q[src + 2 * kp + 1];
    q[src + 2 * kp]     = (bf16_t)(qe * cs - qo * sn);
    q[src + 2 * kp + 1] = (bf16_t)(qo * cs + qe * sn);
    float ke = (float)k[src + 2 * kp], ko = (float)k[src + 2 * kp + 1];
    k[src + 2 * kp]     = (bf16_t)(ke * cs - ko * sn);
    k[src + 2 * kp + 1] = (bf16_t)(ko * cs + ke * sn);

    const size_t vb = ((size_t)b * NUM_HEADS + hh) * HEAD_DIM;
    Vt[(vb + 2 * kp) * SEQ + s]     = v[src + 2 * kp];
    Vt[(vb + 2 * kp + 1) * SEQ + s] = v[src + 2 * kp + 1];
}

// ---------------------------------------------------------------------------
// Flash-style causal attention per (b,h). Block = 128 threads = 4 waves,
// block handles 64 query rows; each wave owns a 16-row tile and the full
// 128-wide O accumulator (8 x v8f). Online softmax in f32, WMMA for QK^T
// and PV. P is re-laid out C-fragment -> A-fragment through private LDS.
// __launch_bounds__(128,1): pin one block per WGP slot so the ~200 live
// VGPRs/lane fit without scratch spills (round-1 disasm showed reload
// traffic in the hot loop).
// ---------------------------------------------------------------------------
__global__ __launch_bounds__(128, 1)
void attn_kernel(const bf16_t* __restrict__ Q,  // [B,S,D] (rope'd)
                 const bf16_t* __restrict__ K,  // [B,S,D] (rope'd)
                 const bf16_t* __restrict__ Vt, // [B,H,HD,S]
                 bf16_t* __restrict__ O) {      // [B,S,D]
    __shared__ __align__(128) bf16_t lds_p[4][16 * 32];

    const int lane = threadIdx.x & 31;
    const int wave = threadIdx.x >> 5;
    const int bh = blockIdx.y;
    const int b = bh >> 4;
    const int h = bh & 15;
    const int q_base = blockIdx.x * 64 + wave * 16;

    const bf16_t* Qh = Q + ((size_t)b * SEQ) * D_MODEL + h * HEAD_DIM;
    const bf16_t* Kh = K + ((size_t)b * SEQ) * D_MODEL + h * HEAD_DIM;
    const bf16_t* Vh = Vt + ((size_t)bh) * HEAD_DIM * SEQ;

    const int row  = lane & 15;
    const int kgrp = (lane >> 4) * 16;
    const int n_lane  = lane & 15;
    const int m_base8 = (lane >> 4) * 8;

    // Q fragments covering the full head_dim (K = 128 -> 4 x 16x32 frags)
    v16bf qf[4];
    #pragma unroll
    for (int i = 0; i < 4; ++i)
        qf[i] = *(const v16bf*)(Qh + (size_t)(q_base + row) * D_MODEL + i * 32 + kgrp);

    v8f o[8];
    #pragma unroll
    for (int d = 0; d < 8; ++d) o[d] = v8f_zero();
    float mrow[8], lrow[8];
    #pragma unroll
    for (int r = 0; r < 8; ++r) { mrow[r] = -3.0e38f; lrow[r] = 0.f; }

    const int q_hi = q_base + 15;
    bf16_t* lp = lds_p[wave];

    for (int kb = 0; kb <= q_hi; kb += 32) {
        // ---- batched K-fragment loads for the 16x32 score tile ----
        v16bf kf[2][4];
        #pragma unroll
        for (int t = 0; t < 2; ++t)
            #pragma unroll
            for (int kk = 0; kk < 4; ++kk)
                kf[t][kk] = *(const v16bf*)(Kh + (size_t)(kb + t * 16 + row) * D_MODEL + kk * 32 + kgrp);

        v8f sc[2];
        sc[0] = v8f_zero(); sc[1] = v8f_zero();
        #pragma unroll
        for (int t = 0; t < 2; ++t)
            #pragma unroll
            for (int kk = 0; kk < 4; ++kk)
                sc[t] = wmma_bf16(qf[kk], kf[t][kk], sc[t]);

        // ---- scale + causal mask + row max, in place ----
        float rm[8];
        #pragma unroll
        for (int r = 0; r < 8; ++r) rm[r] = -3.0e38f;
        #pragma unroll
        for (int t = 0; t < 2; ++t)
            #pragma unroll
            for (int r = 0; r < 8; ++r) {
                float s = sc[t][r] * SM_SCALE;
                int qi = q_base + m_base8 + r;
                int ki = kb + t * 16 + n_lane;
                if (ki > qi) s -= NEG_MASK;     // matches reference subtraction
                sc[t][r] = s;
                rm[r] = fmaxf(rm[r], s);
            }
        #pragma unroll
        for (int off = 1; off < 16; off <<= 1)
            #pragma unroll
            for (int r = 0; r < 8; ++r) rm[r] = fmaxf(rm[r], __shfl_xor(rm[r], off, 32));

        float alpha[8];
        #pragma unroll
        for (int r = 0; r < 8; ++r) {
            float mnew = fmaxf(mrow[r], rm[r]);
            alpha[r] = __expf(mrow[r] - mnew);
            mrow[r] = mnew;
        }

        // ---- P = exp(s - m) in place, partial row sums ----
        float rs[8];
        #pragma unroll
        for (int r = 0; r < 8; ++r) rs[r] = 0.f;
        #pragma unroll
        for (int t = 0; t < 2; ++t)
            #pragma unroll
            for (int r = 0; r < 8; ++r) {
                float p = __expf(sc[t][r] - mrow[r]);
                sc[t][r] = p;
                rs[r] += p;
            }
        #pragma unroll
        for (int off = 1; off < 16; off <<= 1)
            #pragma unroll
            for (int r = 0; r < 8; ++r) rs[r] += __shfl_xor(rs[r], off, 32);
        #pragma unroll
        for (int r = 0; r < 8; ++r) lrow[r] = lrow[r] * alpha[r] + rs[r];

        // ---- rescale O accumulators ----
        #pragma unroll
        for (int d = 0; d < 8; ++d)
            #pragma unroll
            for (int r = 0; r < 8; ++r) o[d][r] *= alpha[r];

        // ---- redistribute P: C-fragment -> A-fragment via private LDS ----
        #pragma unroll
        for (int t = 0; t < 2; ++t)
            #pragma unroll
            for (int r = 0; r < 8; ++r)
                lp[(m_base8 + r) * 32 + t * 16 + n_lane] = (bf16_t)sc[t][r];
        // per-wave DS ops are in-order; load sees the store
        v16bf pa = *(const v16bf*)(lp + row * 32 + kgrp);

        // ---- O += P * V (batched V-fragment loads, then 8 WMMAs) ----
        v16bf vf[8];
        #pragma unroll
        for (int d = 0; d < 8; ++d)
            vf[d] = *(const v16bf*)(Vh + (size_t)(d * 16 + row) * SEQ + kb + kgrp);
        #pragma unroll
        for (int d = 0; d < 8; ++d)
            o[d] = wmma_bf16(pa, vf[d], o[d]);
    }

    // ---- normalize (reciprocal-multiply) and write O back to [B,S,D] ----
    float inv_l[8];
    #pragma unroll
    for (int r = 0; r < 8; ++r) inv_l[r] = 1.0f / lrow[r];
    #pragma unroll
    for (int d = 0; d < 8; ++d)
        #pragma unroll
        for (int r = 0; r < 8; ++r) {
            int qi = q_base + m_base8 + r;
            int di = d * 16 + n_lane;
            O[((size_t)b * SEQ + qi) * D_MODEL + h * HEAD_DIM + di] = (bf16_t)(o[d][r] * inv_l[r]);
        }
}

// ---------------------------------------------------------------------------
// Launcher
// ---------------------------------------------------------------------------
extern "C" void kernel_launch(void* const* d_in, const int* in_sizes, int n_in,
                              void* d_out, int out_size, void* d_ws, size_t ws_size,
                              hipStream_t stream) {
    const float* x  = (const float*)d_in[0];
    const int*   pos = (const int*)d_in[1];
    const float* Wq = (const float*)d_in[2];
    const float* Wk = (const float*)d_in[3];
    const float* Wv = (const float*)d_in[4];
    const float* Wo = (const float*)d_in[5];
    float* out = (float*)d_out;

    char* ws = (char*)d_ws;
    const size_t XB   = (size_t)MROWS * D_MODEL * sizeof(bf16_t);   // 16 MB
    const size_t WB   = (size_t)D_MODEL * D_MODEL * sizeof(bf16_t); //  8 MB
    bf16_t* xb   = (bf16_t*)(ws);
    bf16_t* Wqb  = (bf16_t*)(ws + XB);
    bf16_t* Wkb  = (bf16_t*)(ws + XB + WB);
    bf16_t* Wvb  = (bf16_t*)(ws + XB + 2 * WB);
    bf16_t* Wob  = (bf16_t*)(ws + XB + 3 * WB);
    bf16_t* qbuf = (bf16_t*)(ws + XB + 4 * WB);            // rope'd in place
    bf16_t* kbuf = (bf16_t*)(ws + 2 * XB + 4 * WB);
    bf16_t* vbuf = (bf16_t*)(ws + 3 * XB + 4 * WB);
    bf16_t* Vt   = (bf16_t*)(ws + 4 * XB + 4 * WB);
    bf16_t* Obuf = xb;  // x is dead after the QKV GEMMs; alias

    const int nx = MROWS * D_MODEL;       // 8388608
    const int nw = D_MODEL * D_MODEL;     // 4194304
    f32_to_bf16_kernel<<<nx / 256, 256, 0, stream>>>(x,  xb,  nx);
    f32_to_bf16_kernel<<<nw / 256, 256, 0, stream>>>(Wq, Wqb, nw);
    f32_to_bf16_kernel<<<nw / 256, 256, 0, stream>>>(Wk, Wkb, nw);
    f32_to_bf16_kernel<<<nw / 256, 256, 0, stream>>>(Wv, Wvb, nw);
    f32_to_bf16_kernel<<<nw / 256, 256, 0, stream>>>(Wo, Wob, nw);

    dim3 gg(D_MODEL / 128, MROWS / 64);   // (16, 64)
    gemm_bf16_wmma<bf16_t><<<gg, 128, 0, stream>>>(xb, Wqb, qbuf, MROWS, D_MODEL, D_MODEL);
    gemm_bf16_wmma<bf16_t><<<gg, 128, 0, stream>>>(xb, Wkb, kbuf, MROWS, D_MODEL, D_MODEL);
    gemm_bf16_wmma<bf16_t><<<gg, 128, 0, stream>>>(xb, Wvb, vbuf, MROWS, D_MODEL, D_MODEL);

    const int nr = BATCH * NUM_HEADS * SEQ * (HEAD_DIM / 2); // 4194304
    rope_v_kernel<<<nr / 256, 256, 0, stream>>>(qbuf, kbuf, vbuf, pos, Vt);

    attn_kernel<<<dim3(SEQ / 64, BATCH * NUM_HEADS), 128, 0, stream>>>(qbuf, kbuf, Vt, Obuf);

    gemm_bf16_wmma<float><<<gg, 128, 0, stream>>>(Obuf, Wob, out, MROWS, D_MODEL, D_MODEL);
}
